// Decoder_85547158601802
// MI455X (gfx1250) — matrix operand order repaired and checked by reading, not compiled, not run on hardware
//
#include <hip/hip_runtime.h>
#include <hip/hip_bf16.h>
#include <math.h>

typedef __bf16 bf16_t;
typedef __attribute__((ext_vector_type(16))) __bf16 v16bf;
typedef __attribute__((ext_vector_type(4)))  __bf16 v4bf;
typedef __attribute__((ext_vector_type(8)))  float  v8f;

#define Bb   32
#define Tt   512
#define Hh   1024
#define Ee   300
#define Vv   32000
#define Ss   50
#define K0PAD 2368   // E + 2H = 2348 padded up to multiple of 32
#define K1   2048
#define N4H  4096

// ---------------------------------------------------------------------------
// Fragment-tiled B layout (global memory):
//   element (k, n) lives at  [nt][kt][lane][e]
//   nt = n>>4, kt = k>>5, lane = 16*((k>>4)&1) + (n&15), e = k&15
// Each 32x16 (KxN) tile is 1KB in exact WMMA B operand order; a wave loads a
// tile as lane-contiguous 32B (two clause'd global_load_b128), and the K loop
// streams contiguous tiles.
// ---------------------------------------------------------------------------
__device__ inline void btile_decode(long i, int Ktiles, int& k, int& n)
{
    int  e    = (int)(i & 15);
    int  lane = (int)((i >> 4) & 31);
    long tile = i >> 9;
    int  kt   = (int)(tile % Ktiles);
    int  nt   = (int)(tile / Ktiles);
    k = kt * 32 + ((lane >> 4) << 4) + e;
    n = nt * 16 + (lane & 15);
}

// ---------------------------------------------------------------------------
// WMMA GEMM: C[M x N] = A[M x K] * Btiled[K x N] + bias[n]
// A row-major (f32 or bf16) staged to LDS in fragment order, double-buffered;
// per thread per chunk: ONE coalesced 16B/8B global load + ONE ds_store_b64
// (the fragment permutation maps 4 consecutive K of a row to 4 consecutive
// LDS slots). B fragments load straight global->VGPR. One barrier per chunk.
// Block = 256 threads (8 waves), tile 32(M) x 128(N), K chunk 32.
// M % 32 == 0, N % 128 == 0, K % 32 == 0 at every call site; bias non-null.
// ---------------------------------------------------------------------------
template<bool A_F32, bool OUT_BF16>
__global__ __launch_bounds__(256)
void gemm_wmma(const void* __restrict__ Aptr, int lda,
               const bf16_t* __restrict__ Btiled,
               const float* __restrict__ bias,
               void* __restrict__ Cptr, int ldc,
               int M, int N, int K)
{
    __shared__ __align__(32) bf16_t lsA[2][32 * 32];   // ping-pong A chunk

    const int m0   = blockIdx.y * 32;
    const int t    = threadIdx.x;
    const int lane = t & 31;
    const int wave = t >> 5;
    const int mh   = wave & 1;   // 16-row half of the 32-row tile
    const int ng   = wave >> 1;  // 32-col group = 2 x 16-wide WMMA tiles
    const int lHi  = lane >> 4;
    const int lLo  = lane & 15;

    // per-thread A staging map: row = t>>3, K quad = (t&7)*4
    const int arow  = t >> 3;
    const int kq    = (t & 7) << 2;
    const int alane = 16 * ((kq >> 4) & 1) + (arow & 15);
    const int ae0   = kq & 15;
    const int adst  = (((arow >> 4) * 32 + alane) << 4) + ae0;

    const int Ktiles = K >> 5;
    const int nt0 = blockIdx.x * 8 + ng * 2;
    const bf16_t* pb0 = Btiled + ((long)(nt0 + 0) * Ktiles * 32 + lane) * 16;
    const bf16_t* pb1 = Btiled + ((long)(nt0 + 1) * Ktiles * 32 + lane) * 16;

    v8f acc0 = {};
    v8f acc1 = {};

    auto stageA = [&](int buf, int kc) {
        if (A_F32) {
            const float* src = (const float*)Aptr + (long)(m0 + arow) * lda + kc + kq;
            float4 x = *(const float4*)src;
            v4bf p;
            p[0] = (bf16_t)x.x; p[1] = (bf16_t)x.y;
            p[2] = (bf16_t)x.z; p[3] = (bf16_t)x.w;
            *(v4bf*)&lsA[buf][adst] = p;
        } else {
            const bf16_t* src = (const bf16_t*)Aptr + (long)(m0 + arow) * lda + kc + kq;
            *(v4bf*)&lsA[buf][adst] = *(const v4bf*)src;
        }
    };

    stageA(0, 0);
    __syncthreads();
    int ib = 0;
    for (int kc = 0; kc < K; kc += 32) {
        v16bf a  = *(const v16bf*)&lsA[ib][(mh * 32 + lane) * 16];
        v16bf b0 = *(const v16bf*)pb0;   pb0 += 512;
        v16bf b1 = *(const v16bf*)pb1;   pb1 += 512;
        acc0 = __builtin_amdgcn_wmma_f32_16x16x32_bf16(false, a, false, b0,
                                                       (short)0, acc0, false, false);
        acc1 = __builtin_amdgcn_wmma_f32_16x16x32_bf16(false, a, false, b1,
                                                       (short)0, acc1, false, false);
        if (kc + 32 < K) stageA(ib ^ 1, kc + 32);   // uniform branch, EXEC intact
        ib ^= 1;
        __syncthreads();
    }

    // epilogue: C layout (ISA 7.12.2): VGPR r -> M = r + 8*hi, N = lane&15
    const int c0 = (nt0 + 0) * 16 + lLo;
    const int c1 = (nt0 + 1) * 16 + lLo;
    const float bia0 = bias[c0];
    const float bia1 = bias[c1];
    #pragma unroll
    for (int r = 0; r < 8; ++r) {
        int m = m0 + mh * 16 + 8 * lHi + r;
        float v0 = acc0[r] + bia0;
        float v1 = acc1[r] + bia1;
        if (OUT_BF16) {
            ((bf16_t*)Cptr)[(long)m * ldc + c0] = (bf16_t)v0;
            ((bf16_t*)Cptr)[(long)m * ldc + c1] = (bf16_t)v1;
        } else {
            ((float*)Cptr)[(long)m * ldc + c0] = v0;
            ((float*)Cptr)[(long)m * ldc + c1] = v1;
        }
    }
}

// ---------------------------------------------------------------------------
// one-time prep kernels: build fragment-tiled bf16 B matrices
// ---------------------------------------------------------------------------
__global__ void k_pack_tr(const float* __restrict__ W, bf16_t* __restrict__ Wt,
                          int N, int K, int ldw, int coloff)
{
    long i = (long)blockIdx.x * blockDim.x + threadIdx.x;
    long total = (long)N * K;
    if (i >= total) return;
    int k, n;
    btile_decode(i, K >> 5, k, n);
    Wt[i] = (bf16_t)W[(long)n * ldw + coloff + k];
}

__global__ void k_pack_Wcat0(const float* __restrict__ Wih, const float* __restrict__ Whh,
                             bf16_t* __restrict__ Wt)
{
    long i = (long)blockIdx.x * blockDim.x + threadIdx.x;
    long total = (long)N4H * K0PAD;
    if (i >= total) return;
    int k, n;
    btile_decode(i, K0PAD >> 5, k, n);
    float val = 0.f;
    if (k < Ee + Hh)          val = Wih[(long)n * (Ee + Hh) + k];
    else if (k < Ee + 2 * Hh) val = Whh[(long)n * Hh + (k - (Ee + Hh))];
    Wt[i] = (bf16_t)val;
}

__global__ void k_pack_Wcat1(const float* __restrict__ Wih, const float* __restrict__ Whh,
                             bf16_t* __restrict__ Wt)
{
    long i = (long)blockIdx.x * blockDim.x + threadIdx.x;
    long total = (long)N4H * K1;
    if (i >= total) return;
    int k, n;
    btile_decode(i, K1 >> 5, k, n);
    float val = (k < Hh) ? Wih[(long)n * Hh + k] : Whh[(long)n * Hh + (k - Hh)];
    Wt[i] = (bf16_t)val;
}

__global__ void k_bias2(const float* a, const float* b, float* out, int n)
{
    int i = blockIdx.x * blockDim.x + threadIdx.x;
    if (i < n) out[i] = a[i] + b[i];
}

__global__ void k_zero(float* out, int n)
{
    int i = blockIdx.x * blockDim.x + threadIdx.x;
    if (i < n) out[i] = 0.f;
}

__global__ void k_init_state(const float* __restrict__ eh, const float* __restrict__ ec,
                             float* __restrict__ hs, float* __restrict__ cs)
{
    int i = blockIdx.x * blockDim.x + threadIdx.x;
    if (i >= 2 * Bb * Hh) return;
    int l = i / (Bb * Hh), r = i % (Bb * Hh), b = r / Hh, j = r % Hh;
    const int half = Hh / 2;
    int sl = 2 * l + (j >= half ? 1 : 0);
    int jj = (j >= half) ? (j - half) : j;
    hs[i] = eh[(long)sl * Bb * half + b * half + jj];
    cs[i] = ec[(long)sl * Bb * half + b * half + jj];
}

// ---------------------------------------------------------------------------
// per-step kernels
// ---------------------------------------------------------------------------
__global__ void k_f32_to_bf16(const float* __restrict__ src, bf16_t* __restrict__ dst, int n)
{
    int i = blockIdx.x * blockDim.x + threadIdx.x;
    if (i < n) dst[i] = (bf16_t)src[i];
}

__device__ inline float wave_sum(float x)
{
    #pragma unroll
    for (int o = 16; o > 0; o >>= 1) x += __shfl_xor(x, o, 32);
    return x;
}

__device__ inline float blk_sum(float x, float* sm)
{
    int t = threadIdx.x;
    sm[t] = x; __syncthreads();
    for (int s = blockDim.x >> 1; s > 0; s >>= 1) {
        if (t < s) sm[t] += sm[t + s];
        __syncthreads();
    }
    float r = sm[0]; __syncthreads();
    return r;
}
__device__ inline float blk_max(float x, float* sm)
{
    int t = threadIdx.x;
    sm[t] = x; __syncthreads();
    for (int s = blockDim.x >> 1; s > 0; s >>= 1) {
        if (t < s) sm[t] = fmaxf(sm[t], sm[t + s]);
        __syncthreads();
    }
    float r = sm[0]; __syncthreads();
    return r;
}

// scores[b,t] = sum_h v[h] * tanh(Whid[b,h] + Wenc[b*T+t,h]); one block per (b,t)
// 256 threads x 4 h each = exactly H; wave32 shuffle reduce + 8-partial combine
__global__ __launch_bounds__(256)
void k_scores(const float* __restrict__ Whid, const bf16_t* __restrict__ Wenc,
              const float* __restrict__ v, float* __restrict__ scores)
{
    __shared__ float sm[8];
    int idx = blockIdx.x;
    int b   = idx >> 9;  // / Tt
    int h0  = threadIdx.x << 2;
    float4 wv = *(const float4*)(Whid + b * Hh + h0);
    v4bf   we = *(const v4bf*)(Wenc + (long)idx * Hh + h0);
    float4 vv = *(const float4*)(v + h0);
    float s = vv.x * tanhf(wv.x + (float)we[0])
            + vv.y * tanhf(wv.y + (float)we[1])
            + vv.z * tanhf(wv.z + (float)we[2])
            + vv.w * tanhf(wv.w + (float)we[3]);
    s = wave_sum(s);
    if ((threadIdx.x & 31) == 0) sm[threadIdx.x >> 5] = s;
    __syncthreads();
    if (threadIdx.x == 0) {
        float r = 0.f;
        #pragma unroll
        for (int i = 0; i < 8; ++i) r += sm[i];
        scores[idx] = r;
    }
}

__global__ __launch_bounds__(256)
void k_softmax(const float* __restrict__ scores, float* __restrict__ aw)
{
    __shared__ float sm[256];
    int b = blockIdx.x;
    float m = -3.402823466e38f;
    for (int t = threadIdx.x; t < Tt; t += 256) m = fmaxf(m, scores[b * Tt + t]);
    m = blk_max(m, sm);
    float s = 0.f;
    for (int t = threadIdx.x; t < Tt; t += 256) s += expf(scores[b * Tt + t] - m);
    s = blk_sum(s, sm);
    float inv = 1.f / s;
    for (int t = threadIdx.x; t < Tt; t += 256)
        aw[b * Tt + t] = expf(scores[b * Tt + t] - m) * inv;
}

// ctx[b,h] = sum_t aw[b,t] * enc[b,t,h]; block = (b, 128-wide h chunk)
__global__ __launch_bounds__(128)
void k_ctx(const float* __restrict__ aw, const float* __restrict__ enc,
           float* __restrict__ ctx)
{
    __shared__ float aws[Tt];
    int b = blockIdx.x;
    int h = blockIdx.y * 128 + threadIdx.x;
    for (int t = threadIdx.x; t < Tt; t += 128) aws[t] = aw[b * Tt + t];
    __syncthreads();
    const float* e = enc + (long)b * Tt * Hh + h;
    float acc = 0.f;
    for (int t = 0; t < Tt; ++t) acc += aws[t] * e[(long)t * Hh];
    ctx[b * Hh + h] = acc;
}

// A0[b, :] = bf16([emb | ctx | h0 | zero-pad])
__global__ void k_build_x(const int* __restrict__ question, const float* __restrict__ emb,
                          const float* __restrict__ ctx, const float* __restrict__ hs,
                          bf16_t* __restrict__ A0, int s)
{
    int i = blockIdx.x * blockDim.x + threadIdx.x;
    if (i >= Bb * K0PAD) return;
    int b = i / K0PAD, k = i % K0PAD;
    float val;
    if (k < Ee) {
        int q = question[b * Ss + s];
        val = emb[(long)q * Ee + k];
    } else if (k < Ee + Hh)     val = ctx[b * Hh + (k - Ee)];
    else if (k < Ee + 2 * Hh)   val = hs[b * Hh + (k - Ee - Hh)];  // layer-0 h
    else                        val = 0.f;
    A0[i] = (bf16_t)val;
}

__device__ inline float sigm(float x) { return 1.f / (1.f + expf(-x)); }

// layer 0: update h0/c0, build A1 = bf16([h0n | old h1])
__global__ void k_lstm0(const float* __restrict__ g, float* __restrict__ hs,
                        float* __restrict__ cs, bf16_t* __restrict__ A1)
{
    int i = blockIdx.x * blockDim.x + threadIdx.x;
    if (i >= Bb * Hh) return;
    int b = i / Hh, j = i % Hh;
    const float* gb = g + (long)b * 4 * Hh;
    float ii = gb[j], ff = gb[Hh + j], gg = gb[2 * Hh + j], oo = gb[3 * Hh + j];
    float c  = cs[b * Hh + j];
    float cn = sigm(ff) * c + sigm(ii) * tanhf(gg);
    float hn = sigm(oo) * tanhf(cn);
    cs[b * Hh + j] = cn;
    hs[b * Hh + j] = hn;
    A1[b * K1 + j]      = (bf16_t)hn;
    A1[b * K1 + Hh + j] = (bf16_t)hs[Bb * Hh + b * Hh + j];  // old layer-1 h
}

// layer 1: update h1/c1, emit bf16 h1 for the generator GEMM
__global__ void k_lstm1(const float* __restrict__ g, float* __restrict__ hs,
                        float* __restrict__ cs, bf16_t* __restrict__ h1A)
{
    int i = blockIdx.x * blockDim.x + threadIdx.x;
    if (i >= Bb * Hh) return;
    int b = i / Hh, j = i % Hh;
    const float* gb = g + (long)b * 4 * Hh;
    float ii = gb[j], ff = gb[Hh + j], gg = gb[2 * Hh + j], oo = gb[3 * Hh + j];
    float c  = cs[Bb * Hh + b * Hh + j];
    float cn = sigm(ff) * c + sigm(ii) * tanhf(gg);
    float hn = sigm(oo) * tanhf(cn);
    cs[Bb * Hh + b * Hh + j] = cn;
    hs[Bb * Hh + b * Hh + j] = hn;
    h1A[b * Hh + j] = (bf16_t)hn;
}

// in-place log_softmax over V for out[b, s, :]
__global__ __launch_bounds__(256)
void k_logsoftmax(float* __restrict__ out, int s)
{
    __shared__ float sm[256];
    int b = blockIdx.x;
    float* p = out + ((long)b * Ss + s) * Vv;
    float m = -3.402823466e38f;
    for (int i = threadIdx.x; i < Vv; i += 256) m = fmaxf(m, p[i]);
    m = blk_max(m, sm);
    float sum = 0.f;
    for (int i = threadIdx.x; i < Vv; i += 256) sum += expf(p[i] - m);
    sum = blk_sum(sum, sm);
    float lse = m + logf(sum);
    for (int i = threadIdx.x; i < Vv; i += 256) p[i] -= lse;
}

// ---------------------------------------------------------------------------
extern "C" void kernel_launch(void* const* d_in, const int* in_sizes, int n_in,
                              void* d_out, int out_size, void* d_ws, size_t ws_size,
                              hipStream_t stream)
{
    (void)in_sizes; (void)n_in; (void)out_size; (void)ws_size;
    const float* enc_out = (const float*)d_in[0];
    const float* enc_h   = (const float*)d_in[1];
    const float* enc_c   = (const float*)d_in[2];
    const float* emb     = (const float*)d_in[3];
    const float* attn_W  = (const float*)d_in[4];
    const float* attn_b  = (const float*)d_in[5];
    const float* vvec    = (const float*)d_in[6];
    const float* Wih0    = (const float*)d_in[7];
    const float* Whh0    = (const float*)d_in[8];
    const float* bih0    = (const float*)d_in[9];
    const float* bhh0    = (const float*)d_in[10];
    const float* Wih1    = (const float*)d_in[11];
    const float* Whh1    = (const float*)d_in[12];
    const float* bih1    = (const float*)d_in[13];
    const float* bhh1    = (const float*)d_in[14];
    const float* genW    = (const float*)d_in[15];
    const float* genb    = (const float*)d_in[16];
    const int*   quest   = (const int*)  d_in[17];
    float* out = (float*)d_out;

    // ---- workspace carve-up (deterministic, every call) ----
    char* base = (char*)d_ws;
    size_t off = 0;
    auto carve = [&](size_t bytes) -> char* {
        char* p = base + off;
        off = (off + bytes + 255) & ~(size_t)255;
        return p;
    };
    bf16_t* WhT    = (bf16_t*)carve((size_t)Hh * Hh * 2);      // fragment-tiled
    bf16_t* WeT    = (bf16_t*)carve((size_t)Hh * Hh * 2);      // fragment-tiled
    bf16_t* genWT  = (bf16_t*)carve((size_t)Hh * Vv * 2);      // fragment-tiled
    bf16_t* Wcat0  = (bf16_t*)carve((size_t)K0PAD * N4H * 2);  // fragment-tiled
    bf16_t* Wcat1  = (bf16_t*)carve((size_t)K1 * N4H * 2);     // fragment-tiled
    bf16_t* Wenc   = (bf16_t*)carve((size_t)Bb * Tt * Hh * 2); // row-major
    float*  b0comb = (float*) carve((size_t)N4H * 4);
    float*  b1comb = (float*) carve((size_t)N4H * 4);
    float*  zbias  = (float*) carve((size_t)Hh * 4);
    float*  hstate = (float*) carve((size_t)2 * Bb * Hh * 4);
    float*  cstate = (float*) carve((size_t)2 * Bb * Hh * 4);
    bf16_t* hidA   = (bf16_t*)carve((size_t)Bb * Hh * 2);
    float*  Whid   = (float*) carve((size_t)Bb * Hh * 4);
    float*  scores = (float*) carve((size_t)Bb * Tt * 4);
    float*  aw     = (float*) carve((size_t)Bb * Tt * 4);
    float*  ctx    = (float*) carve((size_t)Bb * Hh * 4);
    bf16_t* A0     = (bf16_t*)carve((size_t)Bb * K0PAD * 2);
    float*  g0     = (float*) carve((size_t)Bb * N4H * 4);
    bf16_t* A1     = (bf16_t*)carve((size_t)Bb * K1 * 2);
    float*  g1     = (float*) carve((size_t)Bb * N4H * 4);
    bf16_t* h1A    = (bf16_t*)carve((size_t)Bb * Hh * 2);

    auto gblk = [](long n, int bs) { return (unsigned)((n + bs - 1) / bs); };

    // ---- one-time prep: pack all B operands into fragment-tiled bf16 ----
    k_pack_tr<<<gblk((long)Hh * Hh, 256), 256, 0, stream>>>(attn_W, WhT, Hh, Hh, 2 * Hh, 0);
    k_pack_tr<<<gblk((long)Hh * Hh, 256), 256, 0, stream>>>(attn_W, WeT, Hh, Hh, 2 * Hh, Hh);
    k_pack_tr<<<gblk((long)Hh * Vv, 256), 256, 0, stream>>>(genW, genWT, Vv, Hh, Hh, 0);
    k_pack_Wcat0<<<gblk((long)K0PAD * N4H, 256), 256, 0, stream>>>(Wih0, Whh0, Wcat0);
    k_pack_Wcat1<<<gblk((long)K1 * N4H, 256), 256, 0, stream>>>(Wih1, Whh1, Wcat1);
    k_bias2<<<gblk(N4H, 256), 256, 0, stream>>>(bih0, bhh0, b0comb, N4H);
    k_bias2<<<gblk(N4H, 256), 256, 0, stream>>>(bih1, bhh1, b1comb, N4H);
    k_zero<<<gblk(Hh, 256), 256, 0, stream>>>(zbias, Hh);
    k_init_state<<<gblk(2 * Bb * Hh, 256), 256, 0, stream>>>(enc_h, enc_c, hstate, cstate);

    // time-invariant attention precompute: Wenc = enc_out @ We.T  (bf16 out)
    gemm_wmma<true, true><<<dim3(Hh / 128, (Bb * Tt) / 32), 256, 0, stream>>>(
        enc_out, Hh, WeT, zbias, Wenc, Hh, Bb * Tt, Hh, Hh);

    // ---- 50 sequential decode steps ----
    for (int s = 0; s < Ss; ++s) {
        k_f32_to_bf16<<<gblk(Bb * Hh, 256), 256, 0, stream>>>(hstate + Bb * Hh, hidA, Bb * Hh);
        gemm_wmma<false, false><<<dim3(Hh / 128, 1), 256, 0, stream>>>(
            hidA, Hh, WhT, attn_b, Whid, Hh, Bb, Hh, Hh);
        k_scores<<<Bb * Tt, 256, 0, stream>>>(Whid, Wenc, vvec, scores);
        k_softmax<<<Bb, 256, 0, stream>>>(scores, aw);
        k_ctx<<<dim3(Bb, Hh / 128), 128, 0, stream>>>(aw, enc_out, ctx);
        k_build_x<<<gblk((long)Bb * K0PAD, 256), 256, 0, stream>>>(quest, emb, ctx, hstate, A0, s);
        gemm_wmma<false, false><<<dim3(N4H / 128, 1), 256, 0, stream>>>(
            A0, K0PAD, Wcat0, b0comb, g0, N4H, Bb, N4H, K0PAD);
        k_lstm0<<<gblk(Bb * Hh, 256), 256, 0, stream>>>(g0, hstate, cstate, A1);
        gemm_wmma<false, false><<<dim3(N4H / 128, 1), 256, 0, stream>>>(
            A1, K1, Wcat1, b1comb, g1, N4H, Bb, N4H, K1);
        k_lstm1<<<gblk(Bb * Hh, 256), 256, 0, stream>>>(g1, hstate, cstate, h1A);
        // generator straight into d_out slice: C[b][n] at out[(b*S+s)*V + n]
        gemm_wmma<false, false><<<dim3(Vv / 128, 1), 256, 0, stream>>>(
            h1A, Hh, genWT, genb, out + (long)s * Vv, Ss * Vv, Bb, Vv, Hh);
        k_logsoftmax<<<Bb, 256, 0, stream>>>(out, s);
    }
}
